// S2ConvNet_deep_20461224198486
// MI455X (gfx1250) — matrix operand order repaired and checked by reading, not compiled, not run on hardware
//
#include <hip/hip_runtime.h>
#include <hip/hip_bf16.h>
#include <math.h>

// ---------------------------------------------------------------------------
// Spherical CNN (S2/SO3 spectral conv net) for MI455X / gfx1250.
//
// Strategy (see analysis): everything is L2-resident (192MB); all dense
// contractions (truncated DFT analysis/synthesis and the per-degree complex
// z-contraction) are cast as f16->f32 WMMA GEMMs using
// v_wmma_f32_16x16x32_f16 (wave32, one wave per 16x16 tile).  Complex GEMM =
// 4 real WMMAs per K-step (2 when A is real); conj(B) folded by negating the
// imaginary B fragment in VGPRs.  Constant tables (Wigner small-d, DH
// quadrature weights, twiddles) are rebuilt on-device every call.
// ---------------------------------------------------------------------------

#define DEV __device__ __forceinline__
typedef _Float16 h16;
typedef __attribute__((ext_vector_type(16))) _Float16 v16h;
typedef __attribute__((ext_vector_type(8)))  _Float16 v8h;
typedef __attribute__((ext_vector_type(8)))  float    v8f;

static __host__ __device__ inline int rup(int x, int a) { return (x + a - 1) / a * a; }

#define PI_D      3.14159265358979323846
#define TWO_PI_F  6.28318530717958647692f

// ------------------------------- WMMA core ---------------------------------

DEV v8f wmma32(v16h a, v16h b, v8f c) {
  // 8 args: (neg_a, A, neg_b, B, c_mod, C, reuse_a, reuse_b)
  return __builtin_amdgcn_wmma_f32_16x16x32_f16(false, a, false, b, (short)0, c,
                                                false, false);
}

// A fragment: 16x32 (MxK).  ISA 7.12.2: lane m=lane&15 holds K =
// {(lane>>4)*8 .. +7} in halves 0..7 and {16+(lane>>4)*8 .. +7} in halves 8..15.
DEV v16h frag_a(const h16* A, int row, int ld, int kk, int lane) {
  const h16* p = A + (size_t)row * ld + kk + ((lane >> 4) << 3);
  v8h lo = *(const v8h*)p;
  v8h hi = *(const v8h*)(p + 16);
  v16h r;
#pragma unroll
  for (int i = 0; i < 8; ++i) { r[i] = lo[i]; r[i + 8] = hi[i]; }
  return r;
}

// B fragment: 32x16 (KxN), loaded from Bt[N][K] row-major.  Lane n=lane&15
// holds K = {(lane>>4)*8 .. +7} in halves 0..7 and {16+(lane>>4)*8 .. +7} in 8..15.
DEV v16h frag_b(const h16* Bt, int row, int ld, int kk, int lane) {
  const h16* p = Bt + (size_t)row * ld + kk + ((lane >> 4) << 3);
  v8h lo = *(const v8h*)p;
  v8h hi = *(const v8h*)(p + 16);
  v16h r;
#pragma unroll
  for (int i = 0; i < 8; ++i) { r[i] = lo[i]; r[i + 8] = hi[i]; }
  return r;
}

// C = A * (CONJ ? conj(B) : B)^T, complex, split re/im f16 planes, f32 acc.
// One wave per 16x16 tile.  All strides multiples of 32 halves (aligned).
template <bool CONJ, bool AREAL>
__global__ __launch_bounds__(32)
void k_cgemm(const h16* __restrict__ Are, const h16* __restrict__ Aim,
             const h16* __restrict__ Bre, const h16* __restrict__ Bim,
             h16* __restrict__ Cre, h16* __restrict__ Cim,
             int Kp, int lda, int ldb, int ldc) {
  const int lane = threadIdx.x & 31;
  const int m0 = blockIdx.x * 16;
  const int n0 = blockIdx.y * 16;
  const int ar = m0 + (lane & 15);
  const int br = n0 + (lane & 15);
  v8f accR = {0.f, 0.f, 0.f, 0.f, 0.f, 0.f, 0.f, 0.f};
  v8f accI = {0.f, 0.f, 0.f, 0.f, 0.f, 0.f, 0.f, 0.f};
  for (int kk = 0; kk < Kp; kk += 32) {
    if (kk + 32 < Kp)  // emits global_prefetch_b8
      __builtin_prefetch(Are + (size_t)ar * lda + kk + 32, 0, 1);
    v16h a_re  = frag_a(Are, ar, lda, kk, lane);
    v16h b_re  = frag_b(Bre, br, ldb, kk, lane);
    v16h b_im  = frag_b(Bim, br, ldb, kk, lane);
    v16h b_imn = -b_im;
    accR = wmma32(a_re, b_re, accR);
    accI = wmma32(a_re, CONJ ? b_imn : b_im, accI);
    if (!AREAL) {
      v16h a_im = frag_a(Aim, ar, lda, kk, lane);
      accR = wmma32(a_im, CONJ ? b_im : b_imn, accR);
      accI = wmma32(a_im, b_re, accI);
    }
  }
  const int r0 = m0 + ((lane >> 4) << 3);
  const int c  = n0 + (lane & 15);
#pragma unroll
  for (int r = 0; r < 8; ++r) {
    Cre[(size_t)(r0 + r) * ldc + c] = (h16)accR[r];
    Cim[(size_t)(r0 + r) * ldc + c] = (h16)accI[r];
  }
}

// --------------------------- table generation ------------------------------

// Wigner small-d d^l_{m',m}(beta) via the closed-form factorial sum (double).
// dh_b>0: beta_j = pi(2j+1)/(4*dh_b) (Driscoll-Healy); else fixed beta.
// out[jb*(2l+1)^2 + mp*(2l+1) + mq], mp = m'+l, mq = m+l.
__global__ void k_wigner(float* out, int l, int nbeta, int dh_b, float beta_fixed) {
  int dim = 2 * l + 1, dd = dim * dim;
  int t = blockIdx.x * blockDim.x + threadIdx.x;
  if (t >= nbeta * dd) return;
  int jb = t / dd, rem = t % dd;
  int mp = rem / dim, mq = rem % dim;
  double beta = (dh_b > 0) ? PI_D * (2 * jb + 1) / (4.0 * dh_b) : (double)beta_fixed;
  int a = mp - l;  // m'
  int b = mq - l;  // m
  double cb = cos(0.5 * beta), sb = sin(0.5 * beta);
  double lpre = 0.5 * (lgamma((double)(l + a) + 1.0) + lgamma((double)(l - a) + 1.0) +
                       lgamma((double)(l + b) + 1.0) + lgamma((double)(l - b) + 1.0));
  int s0 = (b - a) > 0 ? (b - a) : 0;
  int s1 = (l + b) < (l - a) ? (l + b) : (l - a);
  double acc = 0.0;
  for (int s = s0; s <= s1; ++s) {
    double lden = lgamma((double)(l + b - s) + 1.0) + lgamma((double)s + 1.0) +
                  lgamma((double)(a - b + s) + 1.0) + lgamma((double)(l - a - s) + 1.0);
    double v = exp(lpre - lden) * pow(cb, (double)(2 * l + b - a - 2 * s)) *
               pow(sb, (double)(a - b + 2 * s));
    acc += ((a - b + s) & 1) ? -v : v;
  }
  out[t] = (float)acc;
}

// Driscoll-Healy quadrature weights, length 2b.
__global__ void k_dhw(float* w, int b) {
  int j = blockIdx.x * blockDim.x + threadIdx.x;
  if (j >= 2 * b) return;
  double beta = PI_D * (2 * j + 1) / (4.0 * b);
  double s = 0.0;
  for (int k = 0; k < b; ++k) s += sin((2.0 * k + 1.0) * beta) / (2.0 * k + 1.0);
  w[j] = (float)((2.0 / b) * sin(beta) * s);
}

// Truncated DFT twiddle table, stored as Bt[row][col] f16 planes, zero-padded.
// freqIsRow=1 (analysis):  Bt[f][k] = e^{sgn*2pi*i*(f-L)*k/N}
// freqIsRow=0 (synthesis): Bt[j][f] = e^{sgn*2pi*i*j*(f-L)/N}
__global__ void k_dftW(h16* re, h16* im, int ld, int rowsTot, int rowsValid,
                       int colsValid, int N, int L, float sgn, int freqIsRow) {
  int t = blockIdx.x * blockDim.x + threadIdx.x;
  if (t >= rowsTot * ld) return;
  int r = t / ld, c = t % ld;
  float vr = 0.f, vi = 0.f;
  if (r < rowsValid && c < colsValid) {
    float fr = freqIsRow ? (float)(r - L) : (float)(c - L);
    float tm = freqIsRow ? (float)c : (float)r;
    float th = sgn * TWO_PI_F * fr * tm / (float)N;
    vr = cosf(th); vi = sinf(th);
  }
  re[t] = (h16)vr; im[t] = (h16)vi;
}

// ------------------------------ glue kernels -------------------------------

__global__ void k_pack_x(h16* dst, const float* src, int rows, int N, int ld) {
  int t = blockIdx.x * blockDim.x + threadIdx.x;
  if (t >= rows * ld) return;
  int r = t / ld, c = t % ld;
  dst[t] = (h16)((c < N) ? src[r * N + c] : 0.f);
}

// Out-of-place complex transpose of last two dims (per slice).
__global__ void k_tr(h16* dre, h16* dimg, const h16* sre, const h16* simg,
                     int slices, int rows, int cols, int lds, int ldd) {
  size_t t = (size_t)blockIdx.x * blockDim.x + threadIdx.x;
  size_t tot = (size_t)slices * rows * cols;
  if (t >= tot) return;
  int c = (int)(t % cols); size_t q = t / cols;
  int r = (int)(q % rows); int sl = (int)(q / rows);
  size_t si = ((size_t)sl * rows + r) * lds + c;
  size_t di = ((size_t)sl * cols + c) * ldd + r;
  dre[di] = sre[si]; dimg[di] = simg[si];
}

// xh[(b,m),(i,k)] = sum_j w[j]*d_in^l[j,m,k]*xf[b,i,j,fn(k),fm(m)]  (zero-padded)
__global__ void k_xh(h16* Xre, h16* Xim, const h16* xfre, const h16* xfim,
                     const float* din, const float* wq, int l, int Bn, int Ci,
                     int Jb, int F, int Fp, int Lout, int s2, int Mp, int Kp) {
  int t = blockIdx.x * blockDim.x + threadIdx.x;
  if (t >= Mp * Kp) return;
  int r = t / Kp, c = t % Kp;
  int dim = 2 * l + 1, dd = dim * dim;
  int Kvalid = s2 ? Ci : Ci * dim;
  float are = 0.f, aim = 0.f;
  if (r < Bn * dim && c < Kvalid) {
    int b = r / dim, mp = r % dim;
    int i  = s2 ? c : c / dim;
    int kp = s2 ? 0 : c % dim;
    int fm = mp - l + Lout;
    int fn = kp - l + Lout;
    for (int j = 0; j < Jb; ++j) {
      float dv = s2 ? din[j * dd + mp * dim + l] : din[j * dd + mp * dim + kp];
      float wd = wq[j] * dv;
      size_t idx = s2 ? ((size_t)((b * Ci + i) * Jb + j) * Fp + fm)
                      : (((size_t)((b * Ci + i) * Jb + j) * F + fn) * Fp + fm);
      are += wd * (float)xfre[idx];
      aim += wd * (float)xfim[idx];
    }
  }
  Xre[(size_t)r * Kp + c] = (h16)are;
  Xim[(size_t)r * Kp + c] = (h16)aim;
}

// yh[(o,n),(i,k)] = sum_p kw[i,o,p]*psi^l(n,k,p)  (zero-padded; conj in GEMM)
__global__ void k_yh(h16* Yre, h16* Yim, const float* kw, const float* dg, int l,
                     int Ci, int Co, int P, int n_alpha, int n_gamma, float max_g,
                     int s2, int Np, int Kp) {
  int t = blockIdx.x * blockDim.x + threadIdx.x;
  if (t >= Np * Kp) return;
  int rr = t / Kp, c = t % Kp;
  int dim = 2 * l + 1;
  int Kvalid = s2 ? Ci : Ci * dim;
  float are = 0.f, aim = 0.f;
  if (rr < Co * dim && c < Kvalid) {
    int o = rr / dim, np = rr % dim;
    int i  = s2 ? c : c / dim;
    int kp = s2 ? 0 : c % dim;
    float mt = (float)(np - l), nt = (float)(kp - l);
    for (int p = 0; p < P; ++p) {
      int pa = s2 ? p : p / n_gamma;
      float alpha = TWO_PI_F * pa / (float)n_alpha;
      float dv, phase;
      if (s2) { dv = dg[np * dim + l]; phase = -mt * alpha; }
      else {
        int pg = p % n_gamma;
        float gam = (-max_g + pg * (2.f * max_g / n_gamma)) - alpha;
        dv = dg[np * dim + kp];
        phase = -(mt * alpha + nt * gam);
      }
      float kv = kw[(i * Co + o) * P + p];
      float sn, cs; sincosf(phase, &sn, &cs);
      are += kv * dv * cs; aim += kv * dv * sn;
    }
  }
  Yre[(size_t)rr * Kp + c] = (h16)are;
  Yim[(size_t)rr * Kp + c] = (h16)aim;
}

// G[b,o,j,fm,fn] += (2l+1)*dout[j,m,n]*z[(b,m),(o,n)]
__global__ void k_accG(h16* Gre, h16* Gim, const h16* zre, const h16* zim,
                       const float* dout, int l, int Bn, int Co, int Jbo, int F,
                       int Fp, int Lout, int ldz) {
  int dim = 2 * l + 1, dd = dim * dim;
  size_t t = (size_t)blockIdx.x * blockDim.x + threadIdx.x;
  size_t tot = (size_t)Bn * Co * Jbo * dd;
  if (t >= tot) return;
  int np = (int)(t % dim); size_t q = t / dim;
  int mp = (int)(q % dim); q /= dim;
  int j = (int)(q % Jbo);  q /= Jbo;
  int o = (int)(q % Co);   int b = (int)(q / Co);
  int fm = mp - l + Lout, fn = np - l + Lout;
  size_t zi = (size_t)(b * dim + mp) * ldz + (o * dim + np);
  float dv = (float)(2 * l + 1) * dout[j * dd + mp * dim + np];
  size_t gi = (((size_t)(b * Co + o) * Jbo + j) * F + fm) * Fp + fn;
  Gre[gi] = (h16)((float)Gre[gi] + dv * (float)zre[zi]);
  Gim[gi] = (h16)((float)Gim[gi] + dv * (float)zim[zi]);
}

// h[b,o,j,a,g] = relu(Re U[(b,o,j,g),a]); zero-pads cols for next layer's GEMM.
__global__ void k_finish(h16* h, const h16* Ure, int Bn, int Co, int Jbo,
                         int Nout, int ldo) {
  size_t t = (size_t)blockIdx.x * blockDim.x + threadIdx.x;
  size_t rows = (size_t)Bn * Co * Jbo * Nout;
  if (t >= rows * ldo) return;
  int g = (int)(t % ldo); size_t r = t / ldo;
  int a = (int)(r % Nout); size_t boj = r / Nout;
  float v = 0.f;
  if (g < Nout) {
    float u = (float)Ure[(boj * Nout + g) * (size_t)ldo + a];
    v = u > 0.f ? u : 0.f;
  }
  h[t] = (h16)v;
}

// feat[b,c] = sum_j w[j] * mean_{a,g} h[b,c,j,a,g]
__global__ void k_integrate(float* feat, const h16* h, const float* w, int Bn,
                            int Cn, int Jb, int Na, int ldh) {
  int t = blockIdx.x * blockDim.x + threadIdx.x;
  if (t >= Bn * Cn) return;
  int b = t / Cn, c = t % Cn;
  float acc = 0.f;
  for (int j = 0; j < Jb; ++j) {
    size_t rb = ((size_t)(b * Cn + c) * Jb + j) * Na;
    float inner = 0.f;
    for (int a = 0; a < Na; ++a)
      for (int g = 0; g < Na; ++g) inner += (float)h[(rb + a) * ldh + g];
    acc += w[j] * inner / (float)(Na * Na);
  }
  feat[t] = acc;
}

// ------------------------------- MLP head ----------------------------------

__device__ void mlp_stage(float* X, float* Y, int wi, int wo, const float* g,
                          const float* be, const float* w, const float* b,
                          int doRelu, int t, int nth) {
  if (t < wi) {  // batch-norm over batch dim (B=8, population var)
    float mu = 0.f;
    for (int bb = 0; bb < 8; ++bb) mu += X[bb * wi + t];
    mu *= 0.125f;
    float va = 0.f;
    for (int bb = 0; bb < 8; ++bb) { float d = X[bb * wi + t] - mu; va += d * d; }
    va *= 0.125f;
    float sc = g[t] * rsqrtf(va + 1e-5f);
    for (int bb = 0; bb < 8; ++bb) X[bb * wi + t] = sc * (X[bb * wi + t] - mu) + be[t];
  }
  __syncthreads();
  for (int i = t; i < 8 * wo; i += nth) {
    int bb = i / wo, j = i % wo;
    float a = b[j];
    for (int f = 0; f < wi; ++f) a += X[bb * wi + f] * w[f * wo + j];
    Y[i] = doRelu ? (a > 0.f ? a : 0.f) : a;
  }
  __syncthreads();
  for (int i = t; i < 8 * wo; i += nth) X[i] = Y[i];
  __syncthreads();
}

__global__ void k_mlp(float* out, const float* feat, const float* g1,
                      const float* be1, const float* w1, const float* b1,
                      const float* g2, const float* be2, const float* w2,
                      const float* b2, const float* g3, const float* be3,
                      const float* w3, const float* b3) {
  __shared__ float X[512], Y[512];
  int t = threadIdx.x, nth = blockDim.x;
  for (int i = t; i < 512; i += nth) X[i] = feat[i];
  __syncthreads();
  mlp_stage(X, Y, 64, 64, g1, be1, w1, b1, 1, t, nth);
  mlp_stage(X, Y, 64, 32, g2, be2, w2, b2, 1, t, nth);
  mlp_stage(X, Y, 32, 10, g3, be3, w3, b3, 0, t, nth);
  for (int i = t; i < 80; i += nth) out[i] = X[i];
}

// ------------------------------- host driver -------------------------------

extern "C" void kernel_launch(void* const* d_in, const int* in_sizes, int n_in,
                              void* d_out, int out_size, void* d_ws,
                              size_t ws_size, hipStream_t stream) {
  (void)in_sizes; (void)n_in; (void)out_size; (void)ws_size;
  const float* x = (const float*)d_in[0];
  const float* kws[7];
  for (int i = 0; i < 7; ++i) kws[i] = (const float*)d_in[1 + i];
  const float* bnp[12];
  for (int i = 0; i < 12; ++i) bnp[i] = (const float*)d_in[8 + i];
  float* out = (float*)d_out;

  char* base = (char*)d_ws;
  size_t off = 0;
  auto alloc = [&](size_t bytes) -> char* {
    char* p = base + off;
    off = (off + bytes + 255) & ~(size_t)255;
    return p;
  };

  struct LP { int s2, bI, bO, Ci, Co, P, na, ng; float bg, mg; };
  const float PIf = 3.14159265358979f;
  const LP L[7] = {
      {1, 30, 30,  1,  8,  6, 6, 0, PIf / 16, 0.f},
      {0, 30, 15,  8, 16, 36, 6, 6, PIf / 16, 2 * PIf},
      {0, 15, 15, 16, 16, 36, 6, 6, PIf / 8,  2 * PIf},
      {0, 15,  7, 16, 24, 36, 6, 6, PIf / 8,  2 * PIf},
      {0,  7,  7, 24, 24, 36, 6, 6, PIf / 4,  2 * PIf},
      {0,  7,  3, 24, 32, 36, 6, 6, PIf / 4,  2 * PIf},
      {0,  3,  3, 32, 64, 36, 6, 6, PIf / 2,  2 * PIf},
  };
  const int Bn = 8;

  // ---- size the two big ping/pong spectral regions + activation buffers ----
  size_t sp1e = 0, sp2e = 0, hAe = 0, hBe = 0;
  for (int li = 0; li < 7; ++li) {
    const LP& P = L[li];
    int Nin = 2 * P.bI, Nout = 2 * P.bO, F = 2 * P.bO - 1;
    int NinP = rup(Nin, 32), NoutP = rup(Nout, 32), Fp = rup(F, 32);
    int Jbi = Nin, Jbo = Nout;
    size_t R1 = (size_t)Bn * P.Ci * Jbi * (P.s2 ? 1 : Nin);
    size_t R2 = (size_t)Bn * P.Ci * Jbi * F;
    size_t GR = (size_t)Bn * P.Co * Jbo * F;
    size_t TR = (size_t)Bn * P.Co * Jbo * Nout;
    size_t a1 = (R1 + 32) * (size_t)Fp;
    if (!P.s2 && (R2 + 32) * (size_t)Fp > a1) a1 = (R2 + 32) * (size_t)Fp;
    size_t a2 = (GR + 32) * (size_t)NoutP, a3 = (TR + 32) * (size_t)NoutP;
    size_t m1 = a1; if (a2 > m1) m1 = a2; if (a3 > m1) m1 = a3;
    if (m1 > sp1e) sp1e = m1;
    size_t b1v = P.s2 ? 0 : (R2 + 32) * (size_t)NinP;
    size_t b2v = (GR + 32) * (size_t)Fp, b3v = (TR + 32) * (size_t)Fp;
    size_t m2 = b1v; if (b2v > m2) m2 = b2v; if (b3v > m2) m2 = b3v;
    if (m2 > sp2e) sp2e = m2;
    size_t ho = (TR + 32) * (size_t)NoutP;
    if (li % 2 == 0) { if (ho > hAe) hAe = ho; } else { if (ho > hBe) hBe = ho; }
  }
  h16 *SP1re = (h16*)alloc(sp1e * 2), *SP1im = (h16*)alloc(sp1e * 2);
  h16 *SP2re = (h16*)alloc(sp2e * 2), *SP2im = (h16*)alloc(sp2e * 2);
  h16 *hA = (h16*)alloc(hAe * 2), *hB = (h16*)alloc(hBe * 2);
  h16* h0 = (h16*)alloc((size_t)(Bn * 60 + 32) * 64 * 2);
  const size_t ZE = 1u << 19, XE = 1u << 17, YE = 1u << 18;
  h16 *zre = (h16*)alloc(ZE * 2), *zim = (h16*)alloc(ZE * 2);
  h16 *xre = (h16*)alloc(XE * 2), *xim = (h16*)alloc(XE * 2);
  h16 *yre = (h16*)alloc(YE * 2), *yim = (h16*)alloc(YE * 2);
  float* feat = (float*)alloc((size_t)Bn * 64 * 4);
  float* wint = (float*)alloc(64 * 4);

  auto cgemm = [&](int conj, int areal, const h16* Are, const h16* Aim,
                   const h16* Bre, const h16* Bim, h16* Cre, h16* Cim, int M,
                   int N, int Kp, int lda, int ldb, int ldc) {
    dim3 g((unsigned)((M + 15) / 16), (unsigned)((N + 15) / 16));
    if (conj)
      k_cgemm<true, false><<<g, 32, 0, stream>>>(Are, Aim, Bre, Bim, Cre, Cim, Kp, lda, ldb, ldc);
    else if (areal)
      k_cgemm<false, true><<<g, 32, 0, stream>>>(Are, Aim, Bre, Bim, Cre, Cim, Kp, lda, ldb, ldc);
    else
      k_cgemm<false, false><<<g, 32, 0, stream>>>(Are, Aim, Bre, Bim, Cre, Cim, Kp, lda, ldb, ldc);
  };
  auto ew = [&](size_t tot) { return dim3((unsigned)((tot + 255) / 256)); };

  // pack x [8,1,60,60] -> h0 rows=(b,beta), ld 64, zero-padded cols
  { int rows = Bn * 60; k_pack_x<<<ew((size_t)rows * 64), 256, 0, stream>>>(h0, x, rows, 60, 64); }

  h16* hin = h0;
  for (int li = 0; li < 7; ++li) {
    const LP& P = L[li];
    int Nin = 2 * P.bI, Nout = 2 * P.bO, F = 2 * P.bO - 1;
    int NinP = rup(Nin, 32), NoutP = rup(Nout, 32), Fp = rup(F, 32);
    int Jbi = Nin, Jbo = Nout, Lout = P.bO - 1;

    // ---- tables (rebuilt every call; deterministic) ----
    float* wq = (float*)alloc((size_t)Jbi * 4);
    k_dhw<<<(Jbi + 63) / 64, 64, 0, stream>>>(wq, P.bI);
    int WaRows = rup(F, 16), WsRows = rup(Nout, 16);
    h16* WaRe = (h16*)alloc((size_t)WaRows * NinP * 2);
    h16* WaIm = (h16*)alloc((size_t)WaRows * NinP * 2);
    h16* WsRe = (h16*)alloc((size_t)WsRows * Fp * 2);
    h16* WsIm = (h16*)alloc((size_t)WsRows * Fp * 2);
    k_dftW<<<ew((size_t)WaRows * NinP), 256, 0, stream>>>(WaRe, WaIm, NinP, WaRows, F, Nin, Nin, Lout, -1.f, 1);
    k_dftW<<<ew((size_t)WsRows * Fp), 256, 0, stream>>>(WsRe, WsIm, Fp, WsRows, Nout, F, Nout, Lout, 1.f, 0);
    float *din[30], *dou[30], *dgg[30];
    for (int l = 0; l < P.bO; ++l) {
      int dim = 2 * l + 1, dd = dim * dim;
      din[l] = (float*)alloc((size_t)Jbi * dd * 4);
      dou[l] = (float*)alloc((size_t)Jbo * dd * 4);
      dgg[l] = (float*)alloc((size_t)dd * 4);
      k_wigner<<<ew((size_t)Jbi * dd), 256, 0, stream>>>(din[l], l, Jbi, P.bI, 0.f);
      k_wigner<<<ew((size_t)Jbo * dd), 256, 0, stream>>>(dou[l], l, Jbo, P.bO, 0.f);
      k_wigner<<<ew((size_t)dd), 256, 0, stream>>>(dgg[l], l, 1, 0, P.bg);
    }

    // ---- forward truncated DFT (analysis) -> xf in SP1 ----
    if (P.s2) {
      int rows = Bn * P.Ci * Jbi;
      cgemm(0, 1, hin, hin, WaRe, WaIm, SP1re, SP1im, rows, F, NinP, NinP, NinP, Fp);
    } else {
      int R1v = Bn * P.Ci * Jbi * Nin;
      cgemm(0, 1, hin, hin, WaRe, WaIm, SP1re, SP1im, R1v, F, NinP, NinP, NinP, Fp);
      k_tr<<<ew((size_t)Bn * P.Ci * Jbi * Nin * F), 256, 0, stream>>>(
          SP2re, SP2im, SP1re, SP1im, Bn * P.Ci * Jbi, Nin, F, Fp, NinP);
      int R2v = Bn * P.Ci * Jbi * F;
      cgemm(0, 0, SP2re, SP2im, WaRe, WaIm, SP1re, SP1im, R2v, F, NinP, NinP, NinP, Fp);
    }

    // ---- per-degree contractions; accumulate G (in SP2, zeroed) ----
    size_t GRv = (size_t)Bn * P.Co * Jbo * F;
    size_t gbytes = (GRv + 32) * (size_t)Fp * 2;
    (void)hipMemsetAsync(SP2re, 0, gbytes, stream);
    (void)hipMemsetAsync(SP2im, 0, gbytes, stream);
    for (int l = 0; l < P.bO; ++l) {
      int dim = 2 * l + 1;
      int Mz = Bn * dim, Nz = P.Co * dim, Kz = P.s2 ? P.Ci : P.Ci * dim;
      int Mp = rup(Mz, 16), Np = rup(Nz, 16), Kp = rup(Kz, 32);
      k_xh<<<ew((size_t)Mp * Kp), 256, 0, stream>>>(xre, xim, SP1re, SP1im, din[l], wq,
                                                    l, Bn, P.Ci, Jbi, F, Fp, Lout, P.s2, Mp, Kp);
      k_yh<<<ew((size_t)Np * Kp), 256, 0, stream>>>(yre, yim, kws[li], dgg[l], l, P.Ci,
                                                    P.Co, P.P, P.na, P.ng, P.mg, P.s2, Np, Kp);
      cgemm(1, 0, xre, xim, yre, yim, zre, zim, Mz, Nz, Kp, Kp, Kp, Np);
      k_accG<<<ew((size_t)Bn * P.Co * Jbo * dim * dim), 256, 0, stream>>>(
          SP2re, SP2im, zre, zim, dou[l], l, Bn, P.Co, Jbo, F, Fp, Lout, Np);
    }

    // ---- inverse (synthesis) DFT + ReLU -> next activation ----
    cgemm(0, 0, SP2re, SP2im, WsRe, WsIm, SP1re, SP1im, (int)GRv, Nout, Fp, Fp, Fp, NoutP);
    k_tr<<<ew((size_t)Bn * P.Co * Jbo * F * Nout), 256, 0, stream>>>(
        SP2re, SP2im, SP1re, SP1im, Bn * P.Co * Jbo, F, Nout, NoutP, Fp);
    int TRv = Bn * P.Co * Jbo * Nout;
    cgemm(0, 0, SP2re, SP2im, WsRe, WsIm, SP1re, SP1im, TRv, Nout, Fp, Fp, Fp, NoutP);
    h16* hout = (li % 2 == 0) ? hA : hB;
    k_finish<<<ew((size_t)TRv * NoutP), 256, 0, stream>>>(hout, SP1re, Bn, P.Co, Jbo, Nout, NoutP);
    hin = hout;
  }

  // ---- SO3 integrate + MLP head ----
  k_dhw<<<1, 64, 0, stream>>>(wint, 3);
  k_integrate<<<(Bn * 64 + 63) / 64, 64, 0, stream>>>(feat, hin, wint, Bn, 64, 6, 6, 32);
  k_mlp<<<1, 512, 0, stream>>>(out, feat, bnp[0], bnp[1], bnp[2], bnp[3], bnp[4],
                               bnp[5], bnp[6], bnp[7], bnp[8], bnp[9], bnp[10], bnp[11]);
}
